// QuantumKernelMethod_65481071410012
// MI455X (gfx1250) — matrix operand order repaired and checked by reading, not compiled, not run on hardware
//
#include <hip/hip_runtime.h>

// RBF Gram matrix K[i,j] = exp(-||X_i - Y_j||^2) for X,Y = f32[8192,256].
//
// Plan: split each f32 into bf16 hi + lo once (prepass into d_ws), then compute the
// cross-term GEMM with 3 v_wmma_f32_16x16x32_bf16 per tile (hi*hi + hi*lo + lo*hi,
// f32 accumulate; dropped lo*lo term ~2^-18 relative). Row/col norms are computed
// exactly in f32 per block into LDS; fused exp epilogue with non-temporal stores.
// The kernel is output-store bound (~268 MB @ 23.3 TB/s ~= 11.5 us floor); all
// operand traffic (8.4 MB x2 f32 + 16.8 MB bf16) is L2-resident (192 MB L2).

typedef __attribute__((ext_vector_type(16))) __bf16 v16bf;
typedef __attribute__((ext_vector_type(4)))  __bf16 v4bf;
typedef __attribute__((ext_vector_type(8)))  float  v8f;

// ---- native f32 -> bf16 split: hi = bf16_rne(x), lo = bf16_rne(x - hi) ----
__device__ __forceinline__ void split_native(float x, __bf16& h, __bf16& l) {
  __bf16 hb = (__bf16)x;          // RNE truncation
  h = hb;
  l = (__bf16)(x - (float)hb);    // residual is exact in f32
}

// ---- fragment loads from preconverted bf16 (global, L2-resident) ----
// A-fragment (16x32): lane L -> row M = L%16; elems 0..7 = K base+0..7,
// elems 8..15 = K 16+base+0..7, base = (L<16?0:8). p points at (row, k0+base).
union FragBits { v16bf v; uint4 q[2]; };
__device__ __forceinline__ v16bf loadA_bf(const __bf16* __restrict__ p) {
  FragBits f;
  f.q[0] = *(const uint4*)(p);        // 8 bf16
  f.q[1] = *(const uint4*)(p + 16);   // 8 bf16 at K+16
  return f.v;
}
// B-fragment (32x16): lane L -> col N = L%16; elems 0..7 = K base+0..7,
// elems 8..15 = K 16+base+0..7, base = (L<16?0:8). p points at (col, k0+base).
__device__ __forceinline__ v16bf loadB_bf(const __bf16* __restrict__ p) {
  FragBits f;
  f.q[0] = *(const uint4*)(p);
  f.q[1] = *(const uint4*)(p + 16);
  return f.v;
}

// ---- inline-conversion fallback builders (used only if ws is too small) ----
__device__ __forceinline__ void buildA_f32(const float* __restrict__ p, v16bf& hi, v16bf& lo) {
  float4 f0 = *(const float4*)(p +  0);
  float4 f1 = *(const float4*)(p +  4);
  float4 f2 = *(const float4*)(p + 16);
  float4 f3 = *(const float4*)(p + 20);
  float v[16] = { f0.x, f0.y, f0.z, f0.w,  f1.x, f1.y, f1.z, f1.w,
                  f2.x, f2.y, f2.z, f2.w,  f3.x, f3.y, f3.z, f3.w };
#pragma unroll
  for (int i = 0; i < 16; ++i) { __bf16 h, l; split_native(v[i], h, l); hi[i] = h; lo[i] = l; }
}
__device__ __forceinline__ void buildB_f32(const float* __restrict__ p, v16bf& hi, v16bf& lo) {
  float4 f0 = *(const float4*)(p +  0);
  float4 f1 = *(const float4*)(p +  4);
  float4 f2 = *(const float4*)(p + 16);
  float4 f3 = *(const float4*)(p + 20);
  float v[16] = { f0.x, f0.y, f0.z, f0.w,  f1.x, f1.y, f1.z, f1.w,
                  f2.x, f2.y, f2.z, f2.w,  f3.x, f3.y, f3.z, f3.w };
#pragma unroll
  for (int i = 0; i < 16; ++i) { __bf16 h, l; split_native(v[i], h, l); hi[i] = h; lo[i] = l; }
}

// ---- prepass: split f32 array into bf16 hi / lo arrays (4 values per thread) ----
__global__ __launch_bounds__(256) void bf16_split_kernel(
    const float* __restrict__ src, __bf16* __restrict__ hi, __bf16* __restrict__ lo, int n4) {
  int i = blockIdx.x * blockDim.x + threadIdx.x;
  if (i >= n4) return;
  float4 v = ((const float4*)src)[i];
  float a[4] = { v.x, v.y, v.z, v.w };
  v4bf h, l;
#pragma unroll
  for (int j = 0; j < 4; ++j) { __bf16 hb, lb; split_native(a[j], hb, lb); h[j] = hb; l[j] = lb; }
  ((v4bf*)hi)[i] = h;
  ((v4bf*)lo)[i] = l;
}

// ---- main kernel ----
// Block tile 128(M) x 256(N), 256 threads = 8 waves in 2(M) x 4(N), wave tile 64x64.
// Each wave: 4x4 WMMA accum tiles, 48 bf16 WMMAs per k-step, K = 256 in 8 steps.
template <bool PRE>
__global__ __launch_bounds__(256) void rbf_gram_wmma(
    const float* __restrict__ X, const float* __restrict__ Y,
    const __bf16* __restrict__ Xhi, const __bf16* __restrict__ Xlo,
    const __bf16* __restrict__ Yhi, const __bf16* __restrict__ Ylo,
    float* __restrict__ out, int Ntot)
{
  __shared__ float sxn[128];   // ||X_row||^2, exact f32
  __shared__ float syn[256];   // ||Y_col||^2, exact f32

  const int rowBase = blockIdx.y * 128;
  const int colBase = blockIdx.x * 256;
  const int t = threadIdx.x;

  // ---- cooperative exact-f32 norms (operands are L2-resident; negligible cost) ----
  {
    const float4* yp = (const float4*)(Y + (size_t)(colBase + t) * 256);
    float s = 0.f;
#pragma unroll 8
    for (int i = 0; i < 64; ++i) {
      float4 v = yp[i];
      s = fmaf(v.x, v.x, s); s = fmaf(v.y, v.y, s);
      s = fmaf(v.z, v.z, s); s = fmaf(v.w, v.w, s);
    }
    syn[t] = s;
    if (t < 128) {
      const float4* xp = (const float4*)(X + (size_t)(rowBase + t) * 256);
      float sx = 0.f;
#pragma unroll 8
      for (int i = 0; i < 64; ++i) {
        float4 v = xp[i];
        sx = fmaf(v.x, v.x, sx); sx = fmaf(v.y, v.y, sx);
        sx = fmaf(v.z, v.z, sx); sx = fmaf(v.w, v.w, sx);
      }
      sxn[t] = sx;
    }
  }
  __syncthreads();

  const int lane  = t & 31;
  const int wv    = t >> 5;
  const int waveM = wv & 1;        // 0..1
  const int waveN = wv >> 1;       // 0..3
  const int l16   = lane & 15;
  const int hh    = lane >> 4;     // lane half (0/1)
  const int wm    = waveM * 64;
  const int wn    = waveN * 64;

  v8f acc[4][4];
#pragma unroll
  for (int mt = 0; mt < 4; ++mt)
#pragma unroll
    for (int nt = 0; nt < 4; ++nt) acc[mt][nt] = (v8f){};

  // element offsets (identical indexing for f32 and bf16 arrays)
  int aoff[4], boff[4];
#pragma unroll
  for (int mt = 0; mt < 4; ++mt)
    aoff[mt] = (rowBase + wm + mt * 16 + l16) * 256 + hh * 8;
#pragma unroll
  for (int nt = 0; nt < 4; ++nt)
    boff[nt] = (colBase + wn + nt * 16 + l16) * 256 + hh * 8;

  for (int k0 = 0; k0 < 256; k0 += 32) {
    v16bf Ahi[4], Alo[4];
#pragma unroll
    for (int mt = 0; mt < 4; ++mt) {
      if constexpr (PRE) {
        Ahi[mt] = loadA_bf(Xhi + aoff[mt] + k0);
        Alo[mt] = loadA_bf(Xlo + aoff[mt] + k0);
      } else {
        buildA_f32(X + aoff[mt] + k0, Ahi[mt], Alo[mt]);
      }
    }
#pragma unroll
    for (int nt = 0; nt < 4; ++nt) {
      v16bf Bhi, Blo;
      if constexpr (PRE) {
        Bhi = loadB_bf(Yhi + boff[nt] + k0);
        Blo = loadB_bf(Ylo + boff[nt] + k0);
      } else {
        buildB_f32(Y + boff[nt] + k0, Bhi, Blo);
      }
#pragma unroll
      for (int mt = 0; mt < 4; ++mt) {
        // dot(x,y) ~= hi*hi + hi*lo + lo*hi  (lo*lo ~2^-18 rel., dropped)
        acc[mt][nt] = __builtin_amdgcn_wmma_f32_16x16x32_bf16(
            false, Ahi[mt], false, Bhi, (short)0, acc[mt][nt], false, false);
        acc[mt][nt] = __builtin_amdgcn_wmma_f32_16x16x32_bf16(
            false, Ahi[mt], false, Blo, (short)0, acc[mt][nt], false, false);
        acc[mt][nt] = __builtin_amdgcn_wmma_f32_16x16x32_bf16(
            false, Alo[mt], false, Bhi, (short)0, acc[mt][nt], false, false);
      }
    }
  }

  // ---- epilogue: C/D layout: VGPR r, lane L -> M = r + (L<16?0:8), N = L%16 ----
#pragma unroll
  for (int mt = 0; mt < 4; ++mt) {
#pragma unroll
    for (int nt = 0; nt < 4; ++nt) {
      const int N  = colBase + wn + nt * 16 + l16;
      const float y2 = syn[wn + nt * 16 + l16];
#pragma unroll
      for (int r = 0; r < 8; ++r) {
        const int mloc = wm + mt * 16 + r + hh * 8;
        const float x2 = sxn[mloc];
        float sq = fmaf(-2.0f, acc[mt][nt][r], x2 + y2);  // x2 + y2 - 2*xy
        sq = fmaxf(sq, 0.0f);
        // GAMMA = 1; NT store: output is write-once, keep L2 for operands
        __builtin_nontemporal_store(__expf(-sq), &out[(size_t)(rowBase + mloc) * (size_t)Ntot + N]);
      }
    }
  }
}

extern "C" void kernel_launch(void* const* d_in, const int* in_sizes, int n_in,
                              void* d_out, int out_size, void* d_ws, size_t ws_size,
                              hipStream_t stream) {
  (void)n_in; (void)out_size;
  const float* X = (const float*)d_in[0];
  const float* Y = (const float*)d_in[1];
  float* out = (float*)d_out;
  const int D = 256;
  const int M = in_sizes[0] / D;          // 8192
  const int N = in_sizes[1] / D;          // 8192
  const size_t elemsX = (size_t)M * D;
  const size_t elemsY = (size_t)N * D;
  const size_t need = (elemsX + elemsY) * 2 * sizeof(unsigned short);  // hi+lo, 16.8 MB

  dim3 grid(N / 256, M / 128);            // 32 x 64 blocks
  dim3 block(256);

  if (ws_size >= need) {
    __bf16* Xhi = (__bf16*)d_ws;
    __bf16* Xlo = Xhi + elemsX;
    __bf16* Yhi = Xlo + elemsX;
    __bf16* Ylo = Yhi + elemsY;
    const int n4x = (int)(elemsX / 4), n4y = (int)(elemsY / 4);
    bf16_split_kernel<<<(n4x + 255) / 256, block, 0, stream>>>(X, Xhi, Xlo, n4x);
    bf16_split_kernel<<<(n4y + 255) / 256, block, 0, stream>>>(Y, Yhi, Ylo, n4y);
    rbf_gram_wmma<true><<<grid, block, 0, stream>>>(X, Y, Xhi, Xlo, Yhi, Ylo, out, N);
  } else {
    rbf_gram_wmma<false><<<grid, block, 0, stream>>>(X, Y, nullptr, nullptr, nullptr, nullptr, out, N);
  }
}
